// AdaptiveRankChristoffel_12592844112396
// MI455X (gfx1250) — compile-verified, matrix operand order checked by reading, not run on hardware
//
#include <hip/hip_runtime.h>
#include <stdint.h>

// ---------------- problem constants ----------------
#define DIM      512
#define RANK     64
#define H1       32
#define ROWS     32768          // BATCH * SEQ
#define ROWS_PER_BLOCK 64       // 4 waves * 16 rows
#define NBLK     (ROWS / ROWS_PER_BLOCK)   // 512
#define KSTEPS   (DIM / 32)     // 16
#define WPACK_ELEMS (32 * 2 * 32 * 16)     // 32768 bf16 = 64 KB

typedef __attribute__((ext_vector_type(8)))  float  v8f;
typedef __attribute__((ext_vector_type(16))) __bf16 v16bf;
typedef __attribute__((ext_vector_type(4)))  uint32_t u32x4;
typedef __attribute__((ext_vector_type(8)))  int      i32x8;
typedef __attribute__((ext_vector_type(4)))  int      i32x4;

// ---------------- helpers ----------------
__device__ __forceinline__ __bf16 f2bf(float f) {
  return (__bf16)f;                        // native cvt on gfx1250 (RTE)
}

// WMMA A/B 16-bit fragment K index for element e (0..15) in lane-half h (0/1)
__device__ __forceinline__ int kform(int e, int h) {
  return (e & 7) + ((e >> 3) << 4) + (h << 3);
}

__device__ __forceinline__ v8f wmma_bf16(v16bf a, v16bf b, v8f c) {
  return __builtin_amdgcn_wmma_f32_16x16x32_bf16(
      false, a, false, b, (short)0, c, false, false);
}

// build an A fragment (M=16,K=32 tile) from 16 consecutive-ish floats:
// e0..7 from p0[0..7], e8..15 from p0[16..23]  (p0 already offset by 8*h)
__device__ __forceinline__ v16bf packA(const float* p0) {
  v16bf a;
#pragma unroll
  for (int e = 0; e < 8; ++e)  a[e]     = f2bf(p0[e]);
#pragma unroll
  for (int e = 0; e < 8; ++e)  a[8 + e] = f2bf(p0[16 + e]);
  return a;
}

__device__ __forceinline__ v16bf ldfrag(const unsigned short* p) {
  return *(const v16bf*)p;                 // 32B contiguous per lane
}

__device__ __forceinline__ float fast_tanh(float x) {
  float r;
  asm volatile("v_tanh_f32 %0, %1\n\tv_nop"   // TRANS op + hazard slot
               : "=v"(r) : "v"(x));
  return r;
}

// ---------------- kernel 0: pack U, w1, W^T into WMMA-B fragment layout ----
__global__ __launch_bounds__(256)
void arc_pack(const float* __restrict__ U, const float* __restrict__ w1,
              const float* __restrict__ W,
              unsigned short* __restrict__ Upack,
              unsigned short* __restrict__ W1pack,
              unsigned short* __restrict__ Wpack) {
  int idx = blockIdx.x * 256 + threadIdx.x;
  const int NU = KSTEPS * 4 * 32 * 16;     // 32768
  const int NW1 = KSTEPS * 2 * 32 * 16;    // 16384
  const int NW = WPACK_ELEMS;              // 32768
  if (idx < NU) {
    int e = idx & 15, lane = (idx >> 4) & 31, rest = idx >> 9;
    int nt = rest & 3, ks = rest >> 2;
    int K = ks * 32 + kform(e, lane >> 4);
    int N = nt * 16 + (lane & 15);
    Upack[idx] = __builtin_bit_cast(unsigned short, f2bf(U[K * RANK + N]));
  } else if (idx < NU + NW1) {
    int j = idx - NU;
    int e = j & 15, lane = (j >> 4) & 31, rest = j >> 9;
    int nt = rest & 1, ks = rest >> 1;
    int K = ks * 32 + kform(e, lane >> 4);
    int N = nt * 16 + (lane & 15);
    W1pack[j] = __builtin_bit_cast(unsigned short, f2bf(w1[K * H1 + N]));
  } else if (idx < NU + NW1 + NW) {
    int j = idx - NU - NW1;
    int e = j & 15, lane = (j >> 4) & 31, rest = j >> 9;
    int ks = rest & 1, nt = rest >> 1;
    int k = ks * 32 + kform(e, lane >> 4);   // rank index r (B row)
    int n = nt * 16 + (lane & 15);           // output dim d (B col)
    Wpack[j] = __builtin_bit_cast(unsigned short, f2bf(W[n * RANK + k]));
  }
}

// ---------------- kernel 1: proj = v@U (unmasked), complexity-net partial ---
__global__ __launch_bounds__(128)
void arc_k1(const float* __restrict__ v,
            const unsigned short* __restrict__ Upack,
            const unsigned short* __restrict__ W1pack,
            const float* __restrict__ b1, const float* __restrict__ w2,
            const float* __restrict__ b2,
            float* __restrict__ proj, float* __restrict__ partials) {
  const int tid = threadIdx.x;
  const int wave = tid >> 5, lane = tid & 31;
  const int h = lane >> 4, ln = lane & 15;
  const int rowbase = blockIdx.x * ROWS_PER_BLOCK + wave * 16;
  const float* rowptr = v + (size_t)(rowbase + ln) * DIM;

  v8f pacc[4] = {v8f(0.f), v8f(0.f), v8f(0.f), v8f(0.f)};
  v8f hacc[2] = {v8f(0.f), v8f(0.f)};

#pragma unroll 4
  for (int ks = 0; ks < KSTEPS; ++ks) {
    __builtin_prefetch(rowptr + (ks + 2) * 32, 0, 0);   // global_prefetch_b8
    v16bf a = packA(rowptr + ks * 32 + h * 8);
#pragma unroll
    for (int nt = 0; nt < 4; ++nt) {
      v16bf b = ldfrag(Upack + ((ks * 4 + nt) * 32 + lane) * 16);
      pacc[nt] = wmma_bf16(a, b, pacc[nt]);
    }
#pragma unroll
    for (int ht = 0; ht < 2; ++ht) {
      v16bf b = ldfrag(W1pack + ((ks * 2 + ht) * 32 + lane) * 16);
      hacc[ht] = wmma_bf16(a, b, hacc[ht]);
    }
  }

  // store unmasked proj (D layout: VGPR i -> row i+8h, lane -> col)
  float* pout = proj + (size_t)rowbase * RANK;
#pragma unroll
  for (int nt = 0; nt < 4; ++nt)
#pragma unroll
    for (int i = 0; i < 8; ++i)
      pout[(i + 8 * h) * RANK + nt * 16 + ln] = pacc[nt][i];

  // complexity net: r = 0.1 + 0.9*sigmoid(relu(h+b1)@w2 + b2), sum over 16 rows
  const float w2a = w2[ln], w2b = w2[16 + ln];
  const float b1a = b1[ln], b1b = b1[16 + ln];
  const float bb2 = b2[0];
  float sumr = 0.f;
#pragma unroll
  for (int i = 0; i < 8; ++i) {
    float c = fmaxf(hacc[0][i] + b1a, 0.f) * w2a +
              fmaxf(hacc[1][i] + b1b, 0.f) * w2b;
    c += __shfl_xor(c, 1, 16);
    c += __shfl_xor(c, 2, 16);
    c += __shfl_xor(c, 4, 16);
    c += __shfl_xor(c, 8, 16);
    float s = c + bb2;
    sumr += 0.1f + 0.9f / (1.f + __expf(-s));
  }
  sumr += __shfl_xor(sumr, 16, 32);        // add other half's 8 rows

  __shared__ float lsum[4];
  if (lane == 0) lsum[wave] = sumr;
  __syncthreads();
  if (tid == 0)
    partials[blockIdx.x] = (lsum[0] + lsum[1]) + (lsum[2] + lsum[3]);
}

// ---------------- kernel 2: deterministic reduction -> eff_rank -------------
__global__ __launch_bounds__(256)
void arc_k2(const float* __restrict__ partials, int* __restrict__ effp) {
  __shared__ float s[256];
  int t = threadIdx.x;
  s[t] = partials[t] + partials[t + 256];
  __syncthreads();
#pragma unroll
  for (int o = 128; o > 0; o >>= 1) {
    if (t < o) s[t] += s[t + o];
    __syncthreads();
  }
  if (t == 0) {
    float avg = s[0] / (float)ROWS;
    int e = (int)(avg * (float)RANK);      // truncation like int()
    e = e < 4 ? 4 : (e > RANK ? RANK : e);
    *effp = e;
  }
}

// ---------------- kernel 3: mask, norm, sq @ W^T, tanh ----------------------
__global__ __launch_bounds__(128)
void arc_k3(const float* __restrict__ proj,
            const unsigned short* __restrict__ Wpack,
            const int* __restrict__ effp, float* __restrict__ out) {
  const int tid = threadIdx.x;
  const int wave = tid >> 5, lane = tid & 31;
  const int h = lane >> 4, ln = lane & 15;

  __shared__ float tile[ROWS_PER_BLOCK * RANK];   // 16 KB: proj tile, then sq
  __shared__ unsigned short wlds[WPACK_ELEMS];    // 64 KB: packed W^T frags

  // --- TDM: DMA the 64KB packed W^T into LDS (one descriptor, wave 0 only) --
  if (wave == 0) {
    uint64_t ga = (uint64_t)(size_t)Wpack;
    u32x4 g0;
    g0[0] = 1u;                                  // count=1, user descriptor
    g0[1] = (uint32_t)(size_t)wlds;              // lds_addr
    g0[2] = (uint32_t)ga;                        // global_addr[31:0]
    g0[3] = (uint32_t)((ga >> 32) & 0x01FFFFFFu) | 0x80000000u; // addr[56:32], type=2
    i32x8 g1;
    g1[0] = (int)(2u << 16);                     // data_size = 4B elements
    g1[1] = (int)((uint32_t)(WPACK_ELEMS / 2) << 16);  // tensor_dim0[15:0]=16384
    g1[2] = (int)(1u << 16);                     // tensor_dim1 = 1
    g1[3] = (int)((uint32_t)(WPACK_ELEMS / 2) << 16);  // tile_dim0 = 16384
    g1[4] = 1;                                   // tile_dim1 = 1
    g1[5] = (int)(WPACK_ELEMS / 2);              // tensor_dim0_stride = 16384
    g1[6] = 0;
    g1[7] = 0;
    i32x4 z4 = {0, 0, 0, 0};
    i32x8 z8 = {0, 0, 0, 0, 0, 0, 0, 0};
    __builtin_amdgcn_tensor_load_to_lds(g0, g1, z4, z4, z8, 0);
    __builtin_amdgcn_s_wait_tensorcnt(0);
  }

  // --- async copy this block's 16KB proj tile into LDS (all waves) ---------
  {
    uint32_t ldsBase = (uint32_t)(size_t)tile;
    unsigned long long gbase =
        (unsigned long long)(size_t)(proj + (size_t)blockIdx.x * ROWS_PER_BLOCK * RANK);
#pragma unroll
    for (int it = 0; it < 8; ++it) {             // 128 thr * 16B * 8 = 16KB
      uint32_t loff = ldsBase + (uint32_t)(tid + it * 128) * 16u;
      uint32_t goff = (uint32_t)(tid + it * 128) * 16u;
      asm volatile("global_load_async_to_lds_b128 %0, %1, %2 offset:0"
                   :: "v"(loff), "v"(goff), "s"(gbase) : "memory");
    }
    asm volatile("s_wait_asynccnt 0x0" ::: "memory");
  }
  __syncthreads();

  const int eff = *effp;

  // mask + squared values + per-row norm
  float pp[4][8];
  float rs[8];
#pragma unroll
  for (int i = 0; i < 8; ++i) rs[i] = 0.f;
#pragma unroll
  for (int nt = 0; nt < 4; ++nt) {
    int col = nt * 16 + ln;
    float m = (col < eff) ? 1.f : 0.f;
#pragma unroll
    for (int i = 0; i < 8; ++i) {
      float p = tile[(wave * 16 + i + 8 * h) * RANK + col] * m;
      float q = p * p;
      pp[nt][i] = q;
      rs[i] += q;
    }
  }
  float scl[8];
#pragma unroll
  for (int i = 0; i < 8; ++i) {
    float t = rs[i];
    t += __shfl_xor(t, 1, 16);
    t += __shfl_xor(t, 2, 16);
    t += __shfl_xor(t, 4, 16);
    t += __shfl_xor(t, 8, 16);
    scl[i] = 1.f / (1.f + sqrtf(t) + 1e-8f);
  }
  // write sq back into LDS (same row-major layout)
#pragma unroll
  for (int nt = 0; nt < 4; ++nt)
#pragma unroll
    for (int i = 0; i < 8; ++i)
      tile[(wave * 16 + i + 8 * h) * RANK + nt * 16 + ln] = pp[nt][i] * scl[i];
  __syncthreads();

  // A fragments for sq[16 x 64] (two K-steps of 32), transpose via LDS
  const float* sqrow = tile + (wave * 16 + ln) * RANK;
  v16bf A0 = packA(sqrow + 0 * 32 + h * 8);
  v16bf A1 = packA(sqrow + 1 * 32 + h * 8);

  float* orow = out + (size_t)(blockIdx.x * ROWS_PER_BLOCK + wave * 16) * DIM;
#pragma unroll 4
  for (int nt = 0; nt < 32; ++nt) {
    v16bf B0 = ldfrag(wlds + ((nt * 2 + 0) * 32 + lane) * 16);
    v16bf B1 = ldfrag(wlds + ((nt * 2 + 1) * 32 + lane) * 16);
    v8f acc = v8f(0.f);
    acc = wmma_bf16(A0, B0, acc);
    acc = wmma_bf16(A1, B1, acc);
    int d = nt * 16 + ln;
#pragma unroll
    for (int i = 0; i < 8; ++i)
      orow[(i + 8 * h) * DIM + d] = 10.f * fast_tanh(acc[i] * 0.1f);
  }
}

// ---------------- launch ----------------------------------------------------
extern "C" void kernel_launch(void* const* d_in, const int* in_sizes, int n_in,
                              void* d_out, int out_size, void* d_ws, size_t ws_size,
                              hipStream_t stream) {
  (void)in_sizes; (void)n_in; (void)out_size; (void)ws_size;
  const float* v  = (const float*)d_in[0];
  const float* U  = (const float*)d_in[1];
  const float* W  = (const float*)d_in[2];
  const float* w1 = (const float*)d_in[3];
  const float* b1 = (const float*)d_in[4];
  const float* w2 = (const float*)d_in[5];
  const float* b2 = (const float*)d_in[6];
  float* out = (float*)d_out;

  char* ws = (char*)d_ws;
  float*          partials = (float*)ws;                          //   2 KB
  int*            effp     = (int*)(ws + 2048);
  unsigned short* Upack    = (unsigned short*)(ws + 4096);        //  64 KB
  unsigned short* W1pack   = (unsigned short*)(ws + 4096 + 65536);//  32 KB
  unsigned short* Wpack    = (unsigned short*)(ws + 4096 + 98304);//  64 KB
  float*          proj     = (float*)(ws + 4096 + 163840);        //   8 MB

  arc_pack<<<320, 256, 0, stream>>>(U, w1, W, Upack, W1pack, Wpack);
  arc_k1<<<NBLK, 128, 0, stream>>>(v, Upack, W1pack, b1, w2, b2, proj, partials);
  arc_k2<<<1, 256, 0, stream>>>(partials, effp);
  arc_k3<<<NBLK, 128, 0, stream>>>(proj, Wpack, effp, out);
}
